// GATNet_7859790152292
// MI455X (gfx1250) — compile-verified, matrix operand order checked by reading, not run on hardware
//
#include <hip/hip_runtime.h>
#include <hip/hip_bf16.h>

#define NN 50000
#define NE 1600000
#define ET (NE + NN)          // edges + self loops
#define F_IN 512
#define H1 8
#define C1 8
#define HC1 64                // H1*C1
#define C2 40
#define C2P 48                // padded to 3 wmma N-tiles
#define NEG_SLOPE 0.2f
#define NTILES (NN / 16)      // 3125, exact
#define KS1 (F_IN / 32)       // 16 k-steps layer1
#define KS2 2                 // 64/32 k-steps layer2

typedef __attribute__((ext_vector_type(16))) __bf16 v16bf;
typedef __attribute__((ext_vector_type(8)))  float  v8f;

union BF16V {
  v16bf v;
  unsigned short u[16];
  unsigned int   w[8];
  uint4          q[2];
};

__device__ __forceinline__ unsigned short f2bf(float f) {
  unsigned int u = __float_as_uint(f);
  u += 0x7FFFu + ((u >> 16) & 1u);      // round-to-nearest-even
  return (unsigned short)(u >> 16);
}
__device__ __forceinline__ unsigned int pack2bf(float lo, float hi) {
  return (unsigned int)f2bf(lo) | ((unsigned int)f2bf(hi) << 16);
}

__device__ __forceinline__ float lrelu(float x) { return x > 0.0f ? x : NEG_SLOPE * x; }

// float atomic max via monotone int/uint ordering (finite + -inf init safe)
__device__ __forceinline__ void atomicMaxF(float* addr, float val) {
  if (val >= 0.0f) atomicMax((int*)addr, __float_as_int(val));
  else             atomicMin((unsigned int*)addr, __float_as_uint(val));
}

__device__ __forceinline__ void edge_sd(int e, const int* __restrict__ ei, int& s, int& d) {
  if (e < NE) { s = ei[e]; d = ei[NE + e]; } else { s = e - NE; d = s; }
}

// WMMA fragment K-mapping: lane kg = lane>>4, element j -> K = kg*8 + (j<8 ? j : 16+(j-8))
__device__ __forceinline__ int kmap(int kg, int j) {
  return kg * 8 + ((j < 8) ? j : (16 + (j - 8)));
}

// ---------------- weight pre-swizzle into WMMA-fragment order ----------------
// layout: frag[(ks*NT + nt)*32 + lane][16]  (16 contiguous bf16 per lane = 32B)
__global__ void swz_w1(const float* __restrict__ W1, unsigned short* __restrict__ w1s) {
  int t = blockIdx.x * blockDim.x + threadIdx.x;         // KS1*4*32*16 = 32768
  if (t >= KS1 * 4 * 32 * 16) return;
  int j = t & 15, lane = (t >> 4) & 31, nt = (t >> 9) & 3, ks = t >> 11;
  int col = nt * 16 + (lane & 15);
  int K = ks * 32 + kmap(lane >> 4, j);
  w1s[t] = f2bf(W1[(size_t)K * HC1 + col]);
}
__global__ void swz_w2(const float* __restrict__ W2, unsigned short* __restrict__ w2s) {
  int t = blockIdx.x * blockDim.x + threadIdx.x;         // KS2*3*32*16 = 3072
  if (t >= KS2 * 3 * 32 * 16) return;
  int j = t & 15, lane = (t >> 4) & 31, nt = (t >> 9) % 3, ks = t / (3 * 512);
  int col = nt * 16 + (lane & 15);
  int K = ks * 32 + kmap(lane >> 4, j);
  w2s[t] = (col < C2) ? f2bf(W2[(size_t)K * C2 + col]) : (unsigned short)0;
}

// ---------------- GEMM1: h1[N,64] = x[N,512] @ W1[512,64]  (bf16 WMMA, f32 acc)
__global__ __launch_bounds__(256)
void gemm1_wmma(const float* __restrict__ x, const unsigned short* __restrict__ w1s,
                float* __restrict__ h1) {
  const int tile = blockIdx.x * 8 + (threadIdx.x >> 5);
  if (tile >= NTILES) return;                 // wave-uniform: EXEC stays all-1s
  const int lane = threadIdx.x & 31;
  const int r = lane & 15;                    // M (for A) / N (for B,C)
  const int kg = lane >> 4;                   // K-half select
  const int row = tile * 16 + r;
  const BF16V* __restrict__ wfrag = (const BF16V*)w1s;
  v8f acc[4] = {};
  for (int ks = 0; ks < KS1; ++ks) {
    const int kbase = ks * 32 + kg * 8;
    const float4* xr = (const float4*)(x + (size_t)row * F_IN + kbase);
    float4 p0 = xr[0], p1 = xr[1];            // K = kbase .. kbase+7
    float4 p2 = xr[4], p3 = xr[5];            // K = kbase+16 .. kbase+23
    BF16V a;
    a.w[0] = pack2bf(p0.x, p0.y); a.w[1] = pack2bf(p0.z, p0.w);
    a.w[2] = pack2bf(p1.x, p1.y); a.w[3] = pack2bf(p1.z, p1.w);
    a.w[4] = pack2bf(p2.x, p2.y); a.w[5] = pack2bf(p2.z, p2.w);
    a.w[6] = pack2bf(p3.x, p3.y); a.w[7] = pack2bf(p3.z, p3.w);
#pragma unroll
    for (int nt = 0; nt < 4; ++nt) {
      BF16V b = wfrag[(ks * 4 + nt) * 32 + lane];   // 32B contiguous, coalesced
      acc[nt] = __builtin_amdgcn_wmma_f32_16x16x32_bf16(false, a.v, false, b.v,
                                                        (short)0, acc[nt], false, false);
    }
  }
  const int mbase = tile * 16 + kg * 8;       // C/D layout: VGPR v -> M = 8*kg + v
#pragma unroll
  for (int nt = 0; nt < 4; ++nt)
#pragma unroll
    for (int v = 0; v < 8; ++v)
      h1[(size_t)(mbase + v) * HC1 + nt * 16 + r] = acc[nt][v];
}

// ---------------- GEMM2: h2[N,40] = relu_h[N,64] @ W2pad[64,48]
__global__ __launch_bounds__(256)
void gemm2_wmma(const unsigned short* __restrict__ hbf, const unsigned short* __restrict__ w2s,
                float* __restrict__ h2) {
  const int tile = blockIdx.x * 8 + (threadIdx.x >> 5);
  if (tile >= NTILES) return;
  const int lane = threadIdx.x & 31;
  const int r = lane & 15;
  const int kg = lane >> 4;
  const int row = tile * 16 + r;
  const BF16V* __restrict__ wfrag = (const BF16V*)w2s;
  v8f acc[3] = {};
#pragma unroll
  for (int ks = 0; ks < KS2; ++ks) {
    const int kbase = ks * 32 + kg * 8;
    const uint4* ar = (const uint4*)(hbf + (size_t)row * HC1 + kbase);
    BF16V a;
    a.q[0] = ar[0];                            // K = kbase .. kbase+7  (8 bf16 = 16B)
    a.q[1] = ar[2];                            // K = kbase+16 .. kbase+23
#pragma unroll
    for (int nt = 0; nt < 3; ++nt) {
      BF16V b = wfrag[(ks * 3 + nt) * 32 + lane];
      acc[nt] = __builtin_amdgcn_wmma_f32_16x16x32_bf16(false, a.v, false, b.v,
                                                        (short)0, acc[nt], false, false);
    }
  }
  const int mbase = tile * 16 + kg * 8;
#pragma unroll
  for (int nt = 0; nt < 3; ++nt) {
    const int col = nt * 16 + r;
    if (col < C2)
#pragma unroll
      for (int v = 0; v < 8; ++v)
        h2[(size_t)(mbase + v) * C2 + col] = acc[nt][v];
  }
}

// ---------------- per-node attention dots + softmax-state init + accumulator zero
__global__ void node_att1(const float* __restrict__ h1, const float* __restrict__ ats,
                          const float* __restrict__ atd, float* __restrict__ a1s,
                          float* __restrict__ a1d, float* __restrict__ m1,
                          float* __restrict__ den1, float* __restrict__ out1) {
  int t = blockIdx.x * blockDim.x + threadIdx.x;
  if (t >= NN * H1) return;
  int n = t >> 3, h = t & 7;
  const float* hp = h1 + (size_t)n * HC1 + h * C1;
  float s = 0.0f, d = 0.0f;
#pragma unroll
  for (int c = 0; c < C1; ++c) {
    float v = hp[c];
    s += v * ats[h * C1 + c];
    d += v * atd[h * C1 + c];
    out1[(size_t)n * HC1 + h * C1 + c] = 0.0f;
  }
  a1s[t] = s; a1d[t] = d;
  m1[t] = -__builtin_inff(); den1[t] = 0.0f;
}

__global__ void node_att2(const float* __restrict__ h2, const float* __restrict__ ats,
                          const float* __restrict__ atd, float* __restrict__ a2s,
                          float* __restrict__ a2d, float* __restrict__ m2,
                          float* __restrict__ den2, float* __restrict__ out2) {
  int n = blockIdx.x * blockDim.x + threadIdx.x;
  if (n >= NN) return;
  const float* hp = h2 + (size_t)n * C2;
  float s = 0.0f, d = 0.0f;
#pragma unroll
  for (int c = 0; c < C2; ++c) {
    float v = hp[c];
    s += v * ats[c]; d += v * atd[c];
    out2[(size_t)n * C2 + c] = 0.0f;
  }
  a2s[n] = s; a2d[n] = d;
  m2[n] = -__builtin_inff(); den2[n] = 0.0f;
}

// ---------------- layer-1 edge passes (H=8)
__global__ void edge_max1(const int* __restrict__ ei, const float* __restrict__ a1s,
                          const float* __restrict__ a1d, float* __restrict__ m1) {
  int t = blockIdx.x * blockDim.x + threadIdx.x;
  if (t >= ET * H1) return;
  int e = t >> 3, h = t & 7, s, d;
  edge_sd(e, ei, s, d);
  atomicMaxF(&m1[d * H1 + h], lrelu(a1s[s * H1 + h] + a1d[d * H1 + h]));
}
__global__ void edge_sum1(const int* __restrict__ ei, const float* __restrict__ a1s,
                          const float* __restrict__ a1d, const float* __restrict__ m1,
                          float* __restrict__ den1) {
  int t = blockIdx.x * blockDim.x + threadIdx.x;
  if (t >= ET * H1) return;
  int e = t >> 3, h = t & 7, s, d;
  edge_sd(e, ei, s, d);
  float lg = lrelu(a1s[s * H1 + h] + a1d[d * H1 + h]);
  atomicAdd(&den1[d * H1 + h], expf(lg - m1[d * H1 + h]));
}
__global__ void edge_scatter1(const int* __restrict__ ei, const float* __restrict__ a1s,
                              const float* __restrict__ a1d, const float* __restrict__ m1,
                              const float* __restrict__ den1, const float* __restrict__ h1,
                              float* __restrict__ out1) {
  long long t = (long long)blockIdx.x * blockDim.x + threadIdx.x;
  if (t >= (long long)ET * HC1) return;
  int e = (int)(t >> 6), c = (int)(t & 63), h = c >> 3, s, d;
  edge_sd(e, ei, s, d);
  float lg = lrelu(a1s[s * H1 + h] + a1d[d * H1 + h]);
  float w = expf(lg - m1[d * H1 + h]) / den1[d * H1 + h];
  atomicAdd(&out1[(size_t)d * HC1 + c], w * h1[(size_t)s * HC1 + c]);
}

// ---------------- bias + relu + bf16 convert for layer-2 GEMM input
__global__ void relu1_cvt(const float* __restrict__ out1, const float* __restrict__ b1,
                          unsigned short* __restrict__ hbf) {
  int t = blockIdx.x * blockDim.x + threadIdx.x;
  if (t >= NN * HC1) return;
  float v = out1[t] + b1[t & 63];
  hbf[t] = f2bf(v > 0.0f ? v : 0.0f);
}

// ---------------- layer-2 edge passes (H=1, C=40)
__global__ void edge_max2(const int* __restrict__ ei, const float* __restrict__ a2s,
                          const float* __restrict__ a2d, float* __restrict__ m2) {
  int e = blockIdx.x * blockDim.x + threadIdx.x;
  if (e >= ET) return;
  int s, d; edge_sd(e, ei, s, d);
  atomicMaxF(&m2[d], lrelu(a2s[s] + a2d[d]));
}
__global__ void edge_sum2(const int* __restrict__ ei, const float* __restrict__ a2s,
                          const float* __restrict__ a2d, const float* __restrict__ m2,
                          float* __restrict__ den2) {
  int e = blockIdx.x * blockDim.x + threadIdx.x;
  if (e >= ET) return;
  int s, d; edge_sd(e, ei, s, d);
  atomicAdd(&den2[d], expf(lrelu(a2s[s] + a2d[d]) - m2[d]));
}
__global__ void edge_scatter2(const int* __restrict__ ei, const float* __restrict__ a2s,
                              const float* __restrict__ a2d, const float* __restrict__ m2,
                              const float* __restrict__ den2, const float* __restrict__ h2,
                              float* __restrict__ out2) {
  long long t = (long long)blockIdx.x * blockDim.x + threadIdx.x;
  if (t >= (long long)ET * 64) return;
  int e = (int)(t >> 6), c = (int)(t & 63);
  if (c >= C2) return;
  int s, d; edge_sd(e, ei, s, d);
  float w = expf(lrelu(a2s[s] + a2d[d]) - m2[d]) / den2[d];
  atomicAdd(&out2[(size_t)d * C2 + c], w * h2[(size_t)s * C2 + c]);
}

// ---------------- final: bias + log_softmax over 40 classes
__global__ void logsm(const float* __restrict__ out2, const float* __restrict__ b2,
                      float* __restrict__ out) {
  int n = blockIdx.x * blockDim.x + threadIdx.x;
  if (n >= NN) return;
  float v[C2], m = -__builtin_inff();
#pragma unroll
  for (int c = 0; c < C2; ++c) { v[c] = out2[(size_t)n * C2 + c] + b2[c]; m = fmaxf(m, v[c]); }
  float sum = 0.0f;
#pragma unroll
  for (int c = 0; c < C2; ++c) sum += expf(v[c] - m);
  float lse = logf(sum);
#pragma unroll
  for (int c = 0; c < C2; ++c) out[(size_t)n * C2 + c] = v[c] - m - lse;
}

extern "C" void kernel_launch(void* const* d_in, const int* in_sizes, int n_in,
                              void* d_out, int out_size, void* d_ws, size_t ws_size,
                              hipStream_t stream) {
  const float* x   = (const float*)d_in[0];
  const int*   ei  = (const int*)  d_in[1];
  const float* W1  = (const float*)d_in[2];
  const float* as1 = (const float*)d_in[3];
  const float* ad1 = (const float*)d_in[4];
  const float* b1  = (const float*)d_in[5];
  const float* W2  = (const float*)d_in[6];
  const float* as2 = (const float*)d_in[7];
  const float* ad2 = (const float*)d_in[8];
  const float* b2  = (const float*)d_in[9];
  float* out = (float*)d_out;

  char* p = (char*)d_ws;
  auto alloc = [&](size_t bytes) { char* r = p; p += (bytes + 255) & ~(size_t)255; return r; };
  unsigned short* w1s = (unsigned short*)alloc((size_t)KS1 * 4 * 32 * 16 * 2);
  unsigned short* w2s = (unsigned short*)alloc((size_t)KS2 * 3 * 32 * 16 * 2);
  float* h1   = (float*)alloc((size_t)NN * HC1 * 4);
  float* a1s  = (float*)alloc((size_t)NN * H1 * 4);
  float* a1d  = (float*)alloc((size_t)NN * H1 * 4);
  float* m1   = (float*)alloc((size_t)NN * H1 * 4);
  float* den1 = (float*)alloc((size_t)NN * H1 * 4);
  float* out1 = (float*)alloc((size_t)NN * HC1 * 4);
  unsigned short* hbf = (unsigned short*)alloc((size_t)NN * HC1 * 2);
  float* h2   = (float*)alloc((size_t)NN * C2 * 4);
  float* a2s  = (float*)alloc((size_t)NN * 4);
  float* a2d  = (float*)alloc((size_t)NN * 4);
  float* m2   = (float*)alloc((size_t)NN * 4);
  float* den2 = (float*)alloc((size_t)NN * 4);
  float* out2 = (float*)alloc((size_t)NN * C2 * 4);

  const int TB = 256;
  auto nb = [](long long n) { return (unsigned)((n + 255) / 256); };
  const unsigned gemm_blocks = (NTILES + 7) / 8;

  swz_w1<<<nb((long long)KS1 * 4 * 32 * 16), TB, 0, stream>>>(W1, w1s);
  gemm1_wmma<<<gemm_blocks, TB, 0, stream>>>(x, w1s, h1);
  node_att1<<<nb((long long)NN * H1), TB, 0, stream>>>(h1, as1, ad1, a1s, a1d, m1, den1, out1);
  edge_max1<<<nb((long long)ET * H1), TB, 0, stream>>>(ei, a1s, a1d, m1);
  edge_sum1<<<nb((long long)ET * H1), TB, 0, stream>>>(ei, a1s, a1d, m1, den1);
  edge_scatter1<<<nb((long long)ET * HC1), TB, 0, stream>>>(ei, a1s, a1d, m1, den1, h1, out1);
  relu1_cvt<<<nb((long long)NN * HC1), TB, 0, stream>>>(out1, b1, hbf);
  swz_w2<<<nb((long long)KS2 * 3 * 32 * 16), TB, 0, stream>>>(W2, w2s);
  gemm2_wmma<<<gemm_blocks, TB, 0, stream>>>(hbf, w2s, h2);
  node_att2<<<nb(NN), TB, 0, stream>>>(h2, as2, ad2, a2s, a2d, m2, den2, out2);
  edge_max2<<<nb(ET), TB, 0, stream>>>(ei, a2s, a2d, m2);
  edge_sum2<<<nb(ET), TB, 0, stream>>>(ei, a2s, a2d, m2, den2);
  edge_scatter2<<<nb((long long)ET * 64), TB, 0, stream>>>(ei, a2s, a2d, m2, den2, h2, out2);
  logsm<<<nb(NN), TB, 0, stream>>>(out2, b2, out);
}
